// Attention_29231547416680
// MI455X (gfx1250) — matrix operand order repaired and checked by reading, not compiled, not run on hardware
//
#include <hip/hip_runtime.h>
#include <hip/hip_bf16.h>

// ---------------------------------------------------------------------------
// Multi-head attention forward for MI455X (gfx1250), wave32 + WMMA f16.
//   x:(4,2048,512) f32, mask:(1,4,2048,2048) bool(u8), Wq/Wk/Wv:(512,512) f32
//   out:(4,2048,512) f32
// Pipeline: f32->f16 convert -> QKV projection (WMMA, sched-pinned pipeline)
//           -> flash attention (WMMA, unroll-2 pipelined + async-to-LDS mask)
// ---------------------------------------------------------------------------

typedef _Float16 half8  __attribute__((ext_vector_type(8)));
typedef _Float16 half16 __attribute__((ext_vector_type(16)));
typedef float    v8f    __attribute__((ext_vector_type(8)));

#define B_SZ  4
#define N_TOK 2048
#define CH    512
#define NH    8
#define HD    64

static __device__ __forceinline__ half16 cat8(half8 a, half8 b) {
  return __builtin_shufflevector(a, b, 0,1,2,3,4,5,6,7,8,9,10,11,12,13,14,15);
}
static __device__ __forceinline__ v8f wmma16(half16 a, half16 b, v8f c) {
  // v_wmma_f32_16x16x32_f16  D = A*B + C   (f32 accumulate)
  return __builtin_amdgcn_wmma_f32_16x16x32_f16(false, a, false, b, (short)0, c,
                                                false, false);
}

// ----------------------------- f32 -> f16 ----------------------------------
__global__ void cvt_f32_to_f16(const float* __restrict__ in,
                               _Float16* __restrict__ out, int n8) {
  int i = blockIdx.x * blockDim.x + threadIdx.x;
  if (i >= n8) return;
  const float4* p = (const float4*)in + (size_t)i * 2;
  float4 a = p[0], b = p[1];
  half8 h;
  h[0] = (_Float16)a.x; h[1] = (_Float16)a.y; h[2] = (_Float16)a.z; h[3] = (_Float16)a.w;
  h[4] = (_Float16)b.x; h[5] = (_Float16)b.y; h[6] = (_Float16)b.z; h[7] = (_Float16)b.w;
  *((half8*)out + i) = h;
}

// -------------------- QKV projection: y = x @ W^T --------------------------
// One wave -> 16 tokens x 64 channels (= one head). blockIdx.y selects Q/K/V.
// Q,K stored (B,H,N,D); V stored transposed (B,H,D,N) for the PV GEMM.
static __device__ __forceinline__ void load_proj_frags(
    const _Float16* __restrict__ xrow, const _Float16* __restrict__ wrow,
    int k0, half16& af, half16 bf[4]) {
  af = cat8(*(const half8*)(xrow + k0), *(const half8*)(xrow + k0 + 8));
#pragma unroll
  for (int t = 0; t < 4; ++t) {
    const half8* bp = (const half8*)(wrow + (size_t)t * 16 * CH + k0);
    bf[t] = cat8(bp[0], bp[1]);
  }
}

__global__ __launch_bounds__(256) void proj_qkv(
    const _Float16* __restrict__ xh,
    const _Float16* __restrict__ Wqh, const _Float16* __restrict__ Wkh,
    const _Float16* __restrict__ Wvh,
    _Float16* __restrict__ Qh, _Float16* __restrict__ Kh,
    _Float16* __restrict__ Vt) {
  const int lane = threadIdx.x & 31;
  const int wave = threadIdx.x >> 5;
  const int l16  = lane & 15;
  const int hi   = lane >> 4;

  const int slot = blockIdx.x * 8 + wave;  // 0..4095
  const int mt   = slot >> 3;              // 512 row tiles over B*N
  const int head = slot & 7;               // 64-channel quad == one head
  const int m0   = mt << 4;
  const int which = blockIdx.y;            // 0:Q 1:K 2:V

  const _Float16* W = (which == 0) ? Wqh : (which == 1) ? Wkh : Wvh;
  const _Float16* xrow = xh + (size_t)(m0 + l16) * CH + hi * 16;
  const _Float16* wrow = W + (size_t)(head * HD + l16) * CH + hi * 16;

  v8f acc[4] = {{}, {}, {}, {}};

  // rolled loop + ping-pong buffers; schedule pinned with sched_group_barrier
  // so the machine scheduler cannot sink next-step loads onto their consumers.
  // Tail bodies over-read by one 32-k step; that lands in the next workspace
  // region (always mapped), values are never consumed.
  half16 afA, afB;
  half16 bfA[4], bfB[4];
  load_proj_frags(xrow, wrow, 0, afA, bfA);

#pragma clang loop unroll(disable)
  for (int k0 = 0; k0 < CH; k0 += 64) {
    // body 1: prefetch k0+32 into B, consume A
    load_proj_frags(xrow, wrow, k0 + 32, afB, bfB);
    acc[0] = wmma16(afA, bfA[0], acc[0]);
    acc[1] = wmma16(afA, bfA[1], acc[1]);
    acc[2] = wmma16(afA, bfA[2], acc[2]);
    acc[3] = wmma16(afA, bfA[3], acc[3]);
    // body 2: prefetch k0+64 into A (over-reads on last trip), consume B
    load_proj_frags(xrow, wrow, k0 + 64, afA, bfA);
    acc[0] = wmma16(afB, bfB[0], acc[0]);
    acc[1] = wmma16(afB, bfB[1], acc[1]);
    acc[2] = wmma16(afB, bfB[2], acc[2]);
    acc[3] = wmma16(afB, bfB[3], acc[3]);
    // pin the pipeline: loads stay ahead of the WMMA groups that cover them
    __builtin_amdgcn_sched_group_barrier(0x020, 10, 0);  // VMEM reads (B+A of body1)
    __builtin_amdgcn_sched_group_barrier(0x008, 4, 0);   // 4 WMMA on A
    __builtin_amdgcn_sched_group_barrier(0x020, 2, 0);   // VMEM reads (A next)
    __builtin_amdgcn_sched_group_barrier(0x008, 4, 0);   // 4 WMMA on B
  }

  const int b  = m0 >> 11;  // batch
  const int n0 = (m0 & 2047) + hi * 8;
  if (which < 2) {
    _Float16* O = ((which == 0) ? Qh : Kh) +
                  ((size_t)(b * NH + head) * N_TOK + n0) * HD + l16;
#pragma unroll
    for (int r = 0; r < 8; ++r) {
      O[(size_t)r * HD + 0]  = (_Float16)acc[0][r];
      O[(size_t)r * HD + 16] = (_Float16)acc[1][r];
      O[(size_t)r * HD + 32] = (_Float16)acc[2][r];
      O[(size_t)r * HD + 48] = (_Float16)acc[3][r];
    }
  } else {
    // transposed: Vt[b,h,d,n]; 8 consecutive tokens -> one b128 store
    _Float16* O = Vt + (size_t)(b * NH + head) * HD * N_TOK + n0;
#pragma unroll
    for (int t = 0; t < 4; ++t) {
      half8 pk;
#pragma unroll
      for (int r = 0; r < 8; ++r) pk[r] = (_Float16)acc[t][r];
      *(half8*)(O + (size_t)(t * 16 + l16) * N_TOK) = pk;
    }
  }
}

// ------------------------ flash attention ----------------------------------
static __device__ __forceinline__ void load_kfrags(const _Float16* Kbase,
                                                   int k0, int l16, int hi,
                                                   half16 kf[4]) {
  const _Float16* kr0 = Kbase + (size_t)(k0 + l16) * HD + hi * 16;
  const _Float16* kr1 = Kbase + (size_t)(k0 + 16 + l16) * HD + hi * 16;
  kf[0] = cat8(*(const half8*)(kr0),      *(const half8*)(kr0 + 8));
  kf[1] = cat8(*(const half8*)(kr0 + 32), *(const half8*)(kr0 + 40));
  kf[2] = cat8(*(const half8*)(kr1),      *(const half8*)(kr1 + 8));
  kf[3] = cat8(*(const half8*)(kr1 + 32), *(const half8*)(kr1 + 40));
}
static __device__ __forceinline__ void load_vfrags(const _Float16* Vbase,
                                                   int k0, int l16, int hi,
                                                   half16 vf[4]) {
  const _Float16* vr = Vbase + (size_t)l16 * N_TOK + k0 + hi * 16;
#pragma unroll
  for (int t = 0; t < 4; ++t) {
    const _Float16* p = vr + (size_t)t * 16 * N_TOK;
    vf[t] = cat8(*(const half8*)(p), *(const half8*)(p + 8));
  }
}
// Async DMA of one 16x32-byte mask tile into LDS (32 lanes x 16B = 512B).
static __device__ __forceinline__ void mask_async_tile(
    const unsigned char* __restrict__ gsrc, unsigned lds_off) {
  unsigned long long g = (unsigned long long)(uintptr_t)gsrc;
  asm volatile("global_load_async_to_lds_b128 %0, %1, off"
               :: "v"(lds_off), "v"(g) : "memory");
}

// One wave -> one (b,h) and 16 queries; streams keys 32 at a time,
// key loop unrolled x2 so ping-pong buffers need no register copies.
__global__ __launch_bounds__(256) void flash_attn(
    const _Float16* __restrict__ Qh, const _Float16* __restrict__ Kh,
    const _Float16* __restrict__ Vt, const unsigned char* __restrict__ mask,
    float* __restrict__ out) {
  __shared__ __align__(16) _Float16 pbuf[8][16][32];            // P staging
  __shared__ __align__(16) unsigned char mbuf[8][2][16][32];    // mask DMA dbuf

  const int lane = threadIdx.x & 31;
  const int wave = threadIdx.x >> 5;
  const int l16  = lane & 15;
  const int hi   = lane >> 4;

  const int slot = blockIdx.x * 8 + wave;  // 0..4095
  const int qt   = slot & 127;             // 128 query tiles
  const int bh   = slot >> 7;              // 0..31  (b*8+h)
  const int b    = bh >> 3;
  const int h    = bh & 7;
  const int q0   = qt << 4;

  const _Float16* Qbase = Qh + ((size_t)bh * N_TOK + q0) * HD;
  const _Float16* Kbase = Kh + (size_t)bh * N_TOK * HD;
  const _Float16* Vbase = Vt + (size_t)bh * HD * N_TOK;
  // mask DMA source for this wave's lane: row = lane>>1, 16B half = lane&1
  const unsigned char* msrc =
      mask + (size_t)b * N_TOK * N_TOK + (size_t)(q0 + (lane >> 1)) * N_TOK +
      (lane & 1) * 16;
  const unsigned mdst[2] = {
      (unsigned)(uintptr_t)&mbuf[wave][0][lane >> 1][(lane & 1) * 16],
      (unsigned)(uintptr_t)&mbuf[wave][1][lane >> 1][(lane & 1) * 16]};

  // Q fragments, pre-scaled by 1/sqrt(2*HD) so S comes out already scaled
  const _Float16 sc = (_Float16)0.08838834764831845f;  // 1/sqrt(128)
  half16 aq0, aq1;
  {
    const _Float16* qr = Qbase + l16 * HD;
    aq0 = cat8(*(const half8*)(qr + hi * 16),      *(const half8*)(qr + 8 + hi * 16));
    aq1 = cat8(*(const half8*)(qr + 32 + hi * 16), *(const half8*)(qr + 40 + hi * 16));
    aq0 *= sc;
    aq1 *= sc;
  }

  float m_r[8], l_r[8];
#pragma unroll
  for (int r = 0; r < 8; ++r) { m_r[r] = -1e30f; l_r[r] = 0.0f; }
  v8f o0 = {}, o1 = {}, o2 = {}, o3 = {};

  // one 32-key step; kf = this step's K fragments (preloaded),
  // kfn = next step's K fragments (loaded here, consumed next call)
  auto attn_step = [&](int k0, half16 (&kf)[4], half16 (&kfn)[4],
                       int buf) __attribute__((always_inline)) {
    const bool more = (k0 + 32) < N_TOK;
    if (more) mask_async_tile(msrc + k0 + 32, mdst[buf ^ 1]);  // next tile DMA

    // V fragments for THIS step: issued now, consumed after softmax
    half16 vf[4];
    load_vfrags(Vbase, k0, l16, hi, vf);

    // ---- S = (Q/temp) @ K^T for 32 keys (K fragments preloaded) ----
    v8f s0 = {}, s1 = {};
    s0 = wmma16(aq0, kf[0], s0);
    s1 = wmma16(aq0, kf[2], s1);
    s0 = wmma16(aq1, kf[1], s0);
    s1 = wmma16(aq1, kf[3], s1);

    // K fragments for NEXT step: overlap with softmax + PV below
    if (more) load_kfrags(Kbase, k0 + 32, l16, hi, kfn);

    // wait for this step's mask tile (in-order async: <=1 leaves older done)
    if (more) asm volatile("s_wait_asynccnt 0x1" ::: "memory");
    else      asm volatile("s_wait_asynccnt 0x0" ::: "memory");

    // ---- mask + online softmax (row stats live in 16-lane halves) ----
    float p0[8], p1[8], alpha[8];
#pragma unroll
    for (int r = 0; r < 8; ++r) {
      const unsigned char* mp = &mbuf[wave][buf][hi * 8 + r][l16];
      float v0 = mp[0]  ? -1e30f : s0[r];
      float v1 = mp[16] ? -1e30f : s1[r];
      float t = fmaxf(v0, v1);
      t = fmaxf(t, __shfl_xor(t, 1));
      t = fmaxf(t, __shfl_xor(t, 2));
      t = fmaxf(t, __shfl_xor(t, 4));
      t = fmaxf(t, __shfl_xor(t, 8));
      float mn = fmaxf(m_r[r], t);
      float al = __expf(m_r[r] - mn);
      m_r[r] = mn;
      alpha[r] = al;
      float e0 = __expf(v0 - mn);
      float e1 = __expf(v1 - mn);
      p0[r] = e0;
      p1[r] = e1;
      float rs = e0 + e1;
      rs += __shfl_xor(rs, 1);
      rs += __shfl_xor(rs, 2);
      rs += __shfl_xor(rs, 4);
      rs += __shfl_xor(rs, 8);
      l_r[r] = l_r[r] * al + rs;
    }
#pragma unroll
    for (int r = 0; r < 8; ++r) {
      o0[r] *= alpha[r]; o1[r] *= alpha[r]; o2[r] *= alpha[r]; o3[r] *= alpha[r];
    }

    // ---- transpose P (C/D layout) -> A-fragment layout via LDS ----
#pragma unroll
    for (int r = 0; r < 8; ++r) {
      pbuf[wave][hi * 8 + r][l16]      = (_Float16)p0[r];
      pbuf[wave][hi * 8 + r][16 + l16] = (_Float16)p1[r];
    }
    asm volatile("s_wait_dscnt 0" ::: "memory");  // same-wave LDS RAW
    half16 ap = cat8(*(const half8*)(&pbuf[wave][l16][hi * 16]),
                     *(const half8*)(&pbuf[wave][l16][8 + hi * 16]));

    // ---- O += P @ V  (V fragments were issued at step start) ----
    o0 = wmma16(ap, vf[0], o0);
    o1 = wmma16(ap, vf[1], o1);
    o2 = wmma16(ap, vf[2], o2);
    o3 = wmma16(ap, vf[3], o3);
  };

  // prologue: mask DMA + K fragments for k0 = 0
  mask_async_tile(msrc, mdst[0]);
  half16 kfA[4], kfB[4];
  load_kfrags(Kbase, 0, l16, hi, kfA);

  for (int k0 = 0; k0 < N_TOK; k0 += 64) {  // unroll x2: A/B buffer ping-pong
    attn_step(k0,      kfA, kfB, 0);
    attn_step(k0 + 32, kfB, kfA, 1);
  }

  // ---- normalize and write out[b, q, h*64 + d] (f32) ----
  float* obase = out + ((size_t)b * N_TOK + q0 + hi * 8) * CH + h * HD + l16;
#pragma unroll
  for (int r = 0; r < 8; ++r) {
    float inv = 1.0f / l_r[r];
    obase[(size_t)r * CH + 0]  = o0[r] * inv;
    obase[(size_t)r * CH + 16] = o1[r] * inv;
    obase[(size_t)r * CH + 32] = o2[r] * inv;
    obase[(size_t)r * CH + 48] = o3[r] * inv;
  }
}

// ---------------------------------------------------------------------------
extern "C" void kernel_launch(void* const* d_in, const int* in_sizes, int n_in,
                              void* d_out, int out_size, void* d_ws,
                              size_t ws_size, hipStream_t stream) {
  const float* x          = (const float*)d_in[0];
  const unsigned char* mk = (const unsigned char*)d_in[1];  // bool tensor, 1B/elem
  const float* Wq         = (const float*)d_in[2];
  const float* Wk         = (const float*)d_in[3];
  const float* Wv         = (const float*)d_in[4];
  float* out              = (float*)d_out;

  const size_t NX = (size_t)B_SZ * N_TOK * CH;  // 4,194,304
  const size_t NW = (size_t)CH * CH;            //   262,144

  char* ws = (char*)d_ws;
  _Float16* xh  = (_Float16*)ws;                      ws += NX * 2;
  _Float16* wqh = (_Float16*)ws;                      ws += NW * 2;
  _Float16* wkh = (_Float16*)ws;                      ws += NW * 2;
  _Float16* wvh = (_Float16*)ws;                      ws += NW * 2;
  _Float16* Qh  = (_Float16*)ws;                      ws += NX * 2;
  _Float16* Kh  = (_Float16*)ws;                      ws += NX * 2;
  _Float16* Vt  = (_Float16*)ws;                      // (B,H,D,N)

  // 1) f32 -> f16 converts
  cvt_f32_to_f16<<<(int)(NX / 8 + 255) / 256, 256, 0, stream>>>(x, xh, (int)(NX / 8));
  cvt_f32_to_f16<<<(int)(NW / 8 + 255) / 256, 256, 0, stream>>>(Wq, wqh, (int)(NW / 8));
  cvt_f32_to_f16<<<(int)(NW / 8 + 255) / 256, 256, 0, stream>>>(Wk, wkh, (int)(NW / 8));
  cvt_f32_to_f16<<<(int)(NW / 8 + 255) / 256, 256, 0, stream>>>(Wv, wvh, (int)(NW / 8));

  // 2) QKV projection: 4096 wave-tiles per matrix, 8 waves/block, y = Q/K/V
  proj_qkv<<<dim3(512, 3), 256, 0, stream>>>(xh, wqh, wkh, wvh, Qh, Kh, Vt);

  // 3) flash attention: 4096 wave-tiles (32 bh * 128 q-tiles), 8 waves/block
  flash_attn<<<512, 256, 0, stream>>>(Qh, Kh, Vt, mk, out);
}
